// _ObjectAttentionBlock_46712064311320
// MI455X (gfx1250) — matrix operand order repaired and checked by reading, not compiled, not run on hardware
//
#include <hip/hip_runtime.h>
#include <stdint.h>

typedef _Float16 half_t;
typedef __attribute__((ext_vector_type(16))) _Float16 v16h;
typedef __attribute__((ext_vector_type(8)))  _Float16 v8h;
typedef __attribute__((ext_vector_type(8)))  float    v8f;
typedef __attribute__((ext_vector_type(4)))  unsigned int u32x4;
typedef __attribute__((ext_vector_type(8)))  int          i32x8;
typedef __attribute__((ext_vector_type(4)))  int          i32x4;

#define NB    4
#define C_IN  512
#define CKD   256
#define HWPX  16384          // 128*128
#define KP    64
#define NPIX  (NB * HWPX)    // 65536
#define EPS   1e-5f

#if defined(__HIP_DEVICE_COMPILE__) && __has_builtin(__builtin_amdgcn_tensor_load_to_lds) && __has_builtin(__builtin_amdgcn_s_wait_tensorcnt)
#define HAVE_TDM 1
#else
#define HAVE_TDM 0
#endif

// ---------------- WMMA fragment loaders (ISA 7.12.2 layouts) ----------------
// A fragment 16(M)x32(K): lane = M row; lanes 0-15 hold K {0..7,16..23},
// lanes 16-31 hold K {8..15,24..31}. Source row-major [M][stride].
__device__ __forceinline__ v16h frag_a(const half_t* p, int stride) {
  int lane = threadIdx.x & 31;
  const half_t* r = p + (size_t)(lane & 15) * stride + ((lane >> 4) << 3);
  v8h lo = *(const v8h*)r;
  v8h hi = *(const v8h*)(r + 16);
  v16h f;
#pragma unroll
  for (int i = 0; i < 8; ++i) { f[i] = lo[i]; f[i + 8] = hi[i]; }
  return f;
}

// B fragment 32(K)x16(N): lane = N column; lanes 0-15 hold K {0..7,16..23},
// lanes 16-31 hold K {8..15,24..31}. Source stored N-major [N][stride of K].
__device__ __forceinline__ v16h frag_b(const half_t* p, int stride) {
  int lane = threadIdx.x & 31;
  const half_t* r = p + (size_t)(lane & 15) * stride + ((lane >> 4) << 3);
  v8h lo = *(const v8h*)r;
  v8h hi = *(const v8h*)(r + 16);
  v16h f;
#pragma unroll
  for (int i = 0; i < 8; ++i) { f[i] = lo[i]; f[i + 8] = hi[i]; }
  return f;
}

#define WMMA_F16(A, B, C) \
  __builtin_amdgcn_wmma_f32_16x16x32_f16(false, (A), false, (B), (short)0, (C), false, false)

// generic->LDS byte offset (LDS aperture: ADDR[31:0] is the LDS offset)
__device__ __forceinline__ unsigned lds_off(const void* p) {
  return (unsigned)(uintptr_t)p;
}

#if HAVE_TDM
// ---------------- Tensor Data Mover: 2D tile (64 rows x 64 f16) -> LDS ------
// D# per ISA ch.8: group0 = {count,lds_addr,global_addr,type=2},
// group1 = {data_size=2B, tensor dims/strides, tile dims}. 2D -> groups 2/3 zero.
// clang-23 toolchain: 6-arg builtin (g0, g1, g2, g3, extra, cpol).
__device__ __forceinline__ void tdm_load_tile(const half_t* gbase, unsigned ldsByteOff,
                                              int Cin, long long tileByteOff) {
  unsigned long long ga = (unsigned long long)(uintptr_t)gbase + (unsigned long long)tileByteOff;
  unsigned td0 = (unsigned)Cin;                 // tensor_dim0 (elements)
  unsigned td1 = (unsigned)NPIX;                // tensor_dim1
  unsigned long long s0 = (unsigned long long)Cin;  // tensor_dim0_stride
  u32x4 g0;
  g0[0] = 1u;                                   // count=1, user descriptor
  g0[1] = ldsByteOff;                           // lds_addr
  g0[2] = (unsigned)(ga & 0xffffffffu);         // global_addr[31:0]
  g0[3] = (unsigned)((ga >> 32) & 0x01ffffffu) | (2u << 30);  // addr[56:32] | type=2
  i32x8 g1;
  g1[0] = (int)(1u << 16);                      // workgroup_mask=0, data_size=1 (2B)
  g1[1] = (int)((td0 & 0xffffu) << 16);         // tensor_dim0[15:0] @ bits 63:48
  g1[2] = (int)((td0 >> 16) | ((td1 & 0xffffu) << 16));   // td0 hi | td1 lo
  g1[3] = (int)((td1 >> 16) | (64u << 16));     // td1 hi | tile_dim0=64
  g1[4] = (int)64;                              // tile_dim1=64, tile_dim2=0
  g1[5] = (int)(s0 & 0xffffffffu);              // dim0 stride lo
  g1[6] = (int)((s0 >> 32) & 0xffffu);          // dim0 stride hi | dim1 stride lo=0
  g1[7] = 0;
  i32x4 z4 = {0, 0, 0, 0};
  i32x8 z8 = {0, 0, 0, 0, 0, 0, 0, 0};
  __builtin_amdgcn_tensor_load_to_lds(g0, g1, z4, z4, z8, 0);
}
#endif

// ---------------- BN fold: wh[o][c] = w*g/sqrt(v+eps), bias = b - m*s -------
__global__ void fold_bn(const float* __restrict__ w, const float* __restrict__ g,
                        const float* __restrict__ b, const float* __restrict__ m,
                        const float* __restrict__ v, half_t* __restrict__ wh,
                        float* __restrict__ bias, int O, int Cin) {
  int i = blockIdx.x * blockDim.x + threadIdx.x;
  if (i < O) {
    float s = g[i] * rsqrtf(v[i] + EPS);
    bias[i] = b[i] - m[i] * s;
  }
  int stride = gridDim.x * blockDim.x;
  for (int idx = i; idx < O * Cin; idx += stride) {
    int o = idx / Cin;
    float s = g[o] * rsqrtf(v[o] + EPS);
    wh[idx] = (half_t)(w[idx] * s);
  }
}

// ---------------- tiny proxy path (key / value) -----------------------------
__global__ void proxy_cbr(const float* __restrict__ in, const half_t* __restrict__ wh,
                          const float* __restrict__ bias, float* __restrict__ outF,
                          half_t* __restrict__ outH, int transpose, int Cin, int O) {
  int idx = blockIdx.x * blockDim.x + threadIdx.x;
  if (idx >= NB * O * KP) return;
  int k = idx % KP;
  int o = (idx / KP) % O;
  int n = idx / (KP * O);
  const float*  ip = in + (size_t)(n * Cin) * KP + k;
  const half_t* wp = wh + (size_t)o * Cin;
  float acc = bias[o];
  for (int c = 0; c < Cin; ++c) acc += (float)wp[c] * ip[(size_t)c * KP];
  acc = fmaxf(acc, 0.f);
  if (outF) outF[idx] = acc;
  if (outH) {
    if (transpose) outH[(size_t)(n * KP + k) * O + o] = (half_t)acc;   // [n][k][o]
    else           outH[idx] = (half_t)acc;                            // [n][o][k]
  }
}

// ---------------- x: f32 CHW -> f16 HWC (one streaming pass) ----------------
__global__ void __launch_bounds__(256) to_hwc_f16(const float* __restrict__ x,
                                                  half_t* __restrict__ xh) {
  __shared__ half_t T[64][64];
  int pxBase = blockIdx.x * 64;
  int cBase  = blockIdx.y * 64;
  int n  = pxBase / HWPX;
  int hw = pxBase - n * HWPX;
  const float* xb = x + ((size_t)n * C_IN + cBase) * HWPX + hw;
  {
    int cl = threadIdx.x >> 2;
    int p0 = (threadIdx.x & 3) * 16;
    const float* src = xb + (size_t)cl * HWPX + p0;
#pragma unroll
    for (int j = 0; j < 16; ++j) T[p0 + j][cl] = (half_t)src[j];
  }
  __syncthreads();
  {
    int pl = threadIdx.x >> 2;
    int c0 = (threadIdx.x & 3) * 16;
    half_t* dst = xh + (size_t)(pxBase + pl) * C_IN + cBase + c0;
    *(v8h*)dst       = *(const v8h*)&T[pl][c0];
    *(v8h*)(dst + 8) = *(const v8h*)&T[pl][c0 + 8];
  }
}

// ---------------- generic HWC f16 GEMM: out = relu(W @ in^T + bias) ---------
// Block: 256 O-rows x 64 px; B tile (64px x 64c) double-buffered in LDS via
// TDM (fallback: vector loads + ds stores). Each wave: 2 M-subtiles x 4 N.
template <bool OUT_F32_CHW>
__global__ void __launch_bounds__(256) gemm_hwc(const half_t* __restrict__ in,
                                                const half_t* __restrict__ wh,
                                                const float* __restrict__ bias,
                                                half_t* __restrict__ outH,
                                                float* __restrict__ outF,
                                                int Cin, int Ochan) {
  __shared__ half_t BT[2][64 * 64];
  int pxBase = blockIdx.x * 64;
  int oBase  = blockIdx.y * 256;
  int wave = threadIdx.x >> 5, lane = threadIdx.x & 31;
  int mRow[2] = {oBase + wave * 16, oBase + 128 + wave * 16};

  v8f acc[2][4];
#pragma unroll
  for (int mi = 0; mi < 2; ++mi)
#pragma unroll
    for (int t = 0; t < 4; ++t)
#pragma unroll
      for (int r = 0; r < 8; ++r)
        acc[mi][t][r] = bias[mRow[mi] + r + ((lane >> 4) << 3)];

  int nChunks = Cin >> 6;   // K chunks of 64
#if HAVE_TDM
  if (wave == 0)
    tdm_load_tile(in, lds_off(&BT[0][0]), Cin, (long long)pxBase * Cin * 2);
#endif
  for (int i = 0; i < nChunks; ++i) {
    int cur = i & 1;
    int cBase = i * 64;
#if HAVE_TDM
    if (wave == 0) {
      if (i + 1 < nChunks) {
        tdm_load_tile(in, lds_off(&BT[1 - cur][0]), Cin,
                      ((long long)pxBase * Cin + (long long)(i + 1) * 64) * 2);
        __builtin_amdgcn_s_wait_tensorcnt(1);   // chunk i complete, i+1 in flight
      } else {
        __builtin_amdgcn_s_wait_tensorcnt(0);   // last chunk complete
      }
    }
    __syncthreads();
#else
    {   // fallback staging: 256 threads copy 64x64 halves
      int row = threadIdx.x >> 2;
      int c0  = (threadIdx.x & 3) * 16;
      const half_t* s = in + (size_t)(pxBase + row) * Cin + cBase + c0;
      *(v8h*)&BT[cur][row * 64 + c0]     = *(const v8h*)s;
      *(v8h*)&BT[cur][row * 64 + c0 + 8] = *(const v8h*)(s + 8);
    }
    __syncthreads();
#endif
    if (i + 1 < nChunks)
      __builtin_prefetch(wh + (size_t)mRow[0] * Cin + cBase + 64, 0, 1);
#pragma unroll
    for (int ks = 0; ks < 2; ++ks) {
      v16h a0 = frag_a(wh + (size_t)mRow[0] * Cin + cBase + ks * 32, Cin);
      v16h a1 = frag_a(wh + (size_t)mRow[1] * Cin + cBase + ks * 32, Cin);
#pragma unroll
      for (int t = 0; t < 4; ++t) {
        v16h bf = frag_b(&BT[cur][(t * 16) * 64 + ks * 32], 64);
        acc[0][t] = WMMA_F16(a0, bf, acc[0][t]);
        acc[1][t] = WMMA_F16(a1, bf, acc[1][t]);
      }
    }
    __syncthreads();
  }

  if (OUT_F32_CHW) {
    int n  = pxBase / HWPX;
    int hw = pxBase - n * HWPX;
#pragma unroll
    for (int mi = 0; mi < 2; ++mi)
#pragma unroll
      for (int t = 0; t < 4; ++t)
#pragma unroll
        for (int r = 0; r < 8; ++r) {
          int oRow = mRow[mi] + r + ((lane >> 4) << 3);
          outF[((size_t)n * Ochan + oRow) * HWPX + hw + t * 16 + (lane & 15)] =
              fmaxf(acc[mi][t][r], 0.f);
        }
  } else {
#pragma unroll
    for (int mi = 0; mi < 2; ++mi)
#pragma unroll
      for (int t = 0; t < 4; ++t) {
        int px = pxBase + t * 16 + (lane & 15);
#pragma unroll
        for (int r = 0; r < 8; ++r) {
          int oRow = mRow[mi] + r + ((lane >> 4) << 3);
          outH[(size_t)px * Ochan + oRow] = (half_t)fmaxf(acc[mi][t][r], 0.f);
        }
      }
  }
}

// ---------------- attention: sim -> softmax -> ctx --------------------------
// Q f16 [NPIX][256]; kf f16 [NB][64][256]; vf f16 [NB][256][64];
// ctx f16 [NPIX][256]. One block = 128 pixels, 8 waves x 16 rows.
__global__ void __launch_bounds__(256) attn(const half_t* __restrict__ Q,
                                            const half_t* __restrict__ kf,
                                            const half_t* __restrict__ vf,
                                            half_t* __restrict__ ctx) {
  __shared__ half_t attl[8][16 * 64];   // per-wave att tile [16 rows][64 k]
  int tileBase = blockIdx.x * 128;
  int n = tileBase / HWPX;
  int wave = threadIdx.x >> 5, lane = threadIdx.x & 31;
  int mBase = tileBase + wave * 16;
  const half_t* kfn = kf + (size_t)n * KP * CKD;   // [k][c]
  const half_t* vfn = vf + (size_t)n * CKD * KP;   // [c][k]

  v8f s[4];
#pragma unroll
  for (int t = 0; t < 4; ++t)
#pragma unroll
    for (int r = 0; r < 8; ++r) s[t][r] = 0.f;

  for (int cBase = 0; cBase < CKD; cBase += 32) {
    v16h a = frag_a(Q + (size_t)mBase * CKD + cBase, CKD);
#pragma unroll
    for (int t = 0; t < 4; ++t) {
      v16h bf = frag_b(kfn + (size_t)(t * 16) * CKD + cBase, CKD);
      s[t] = WMMA_F16(a, bf, s[t]);
    }
  }
  // scale by CK^-0.5 = 1/16, then row-softmax over 64 (cols t*16 + lane%16)
#pragma unroll
  for (int r = 0; r < 8; ++r) {
    float mx = -3.0e38f;
#pragma unroll
    for (int t = 0; t < 4; ++t) { s[t][r] *= 0.0625f; mx = fmaxf(mx, s[t][r]); }
#pragma unroll
    for (int d = 1; d < 16; d <<= 1) mx = fmaxf(mx, __shfl_xor(mx, d, 32));
    float sum = 0.f;
#pragma unroll
    for (int t = 0; t < 4; ++t) {
      float e = __expf(s[t][r] - mx);
      s[t][r] = e;
      sum += e;
    }
#pragma unroll
    for (int d = 1; d < 16; d <<= 1) sum += __shfl_xor(sum, d, 32);
    float rinv = 1.f / sum;
#pragma unroll
    for (int t = 0; t < 4; ++t)
      attl[wave][(r + ((lane >> 4) << 3)) * 64 + t * 16 + (lane & 15)] =
          (half_t)(s[t][r] * rinv);
  }
  __syncthreads();
  // ctx = att(16x64) @ vf(64x256): A frags from LDS, 16 N-subtiles of 16
  v16h a0 = frag_a(&attl[wave][0], 64);        // K = 0..31
  v16h a1 = frag_a(&attl[wave][0] + 32, 64);   // K = 32..63
  half_t* cb = ctx + (size_t)mBase * CKD;
#pragma unroll
  for (int nt = 0; nt < 16; ++nt) {
    v8f o;
#pragma unroll
    for (int r = 0; r < 8; ++r) o[r] = 0.f;
    v16h b0 = frag_b(vfn + (size_t)(nt * 16) * KP, KP);
    v16h b1 = frag_b(vfn + (size_t)(nt * 16) * KP + 32, KP);
    o = WMMA_F16(a0, b0, o);
    o = WMMA_F16(a1, b1, o);
    int col = nt * 16 + (lane & 15);
#pragma unroll
    for (int r = 0; r < 8; ++r) {
      int m = r + ((lane >> 4) << 3);
      cb[(size_t)m * CKD + col] = (half_t)o[r];
    }
  }
}

// ---------------- host side -------------------------------------------------
extern "C" void kernel_launch(void* const* d_in, const int* in_sizes, int n_in,
                              void* d_out, int out_size, void* d_ws, size_t ws_size,
                              hipStream_t stream) {
  (void)in_sizes; (void)n_in; (void)out_size; (void)ws_size;
  const float* x     = (const float*)d_in[0];
  const float* proxy = (const float*)d_in[1];
  // blocks: fp1, fp2, fo1, fo2, fd, fu -> 5 tensors each starting at d_in[2]
  const float* W[6], *G[6], *Bp[6], *M[6], *V[6];
  for (int i = 0; i < 6; ++i) {
    W[i]  = (const float*)d_in[2 + 5 * i + 0];
    G[i]  = (const float*)d_in[2 + 5 * i + 1];
    Bp[i] = (const float*)d_in[2 + 5 * i + 2];
    M[i]  = (const float*)d_in[2 + 5 * i + 3];
    V[i]  = (const float*)d_in[2 + 5 * i + 4];
  }
  const int Od[6] = {CKD, CKD, CKD, CKD, CKD, C_IN};
  const int Cd[6] = {C_IN, CKD, C_IN, CKD, C_IN, CKD};

  char* ws = (char*)d_ws;
  size_t off = 0;
  auto take = [&](size_t bytes) -> void* {
    off = (off + 255) & ~(size_t)255;
    void* p = ws + off;
    off += bytes;
    return p;
  };
  half_t* wh[6];
  float*  bias[6];
  for (int i = 0; i < 6; ++i) {
    wh[i]   = (half_t*)take((size_t)Od[i] * Cd[i] * sizeof(half_t));
    bias[i] = (float*)take((size_t)Od[i] * sizeof(float));
  }
  float*  t1   = (float*)take((size_t)NB * CKD * KP * sizeof(float));
  half_t* kf_t = (half_t*)take((size_t)NB * KP * CKD * sizeof(half_t));
  half_t* vf_t = (half_t*)take((size_t)NB * CKD * KP * sizeof(half_t));
  half_t* xh   = (half_t*)take((size_t)NPIX * C_IN * sizeof(half_t)); // x f16 HWC
  half_t* bufA = (half_t*)take((size_t)NPIX * CKD * sizeof(half_t));  // H1, then ctx
  half_t* bufB = xh;   // Q reuses xh (x f16 is dead after fp1)

  // 1) fold BN into f16 weights + f32 bias
  for (int i = 0; i < 6; ++i) {
    int total = Od[i] * Cd[i];
    fold_bn<<<dim3((total + 255) / 256), dim3(256), 0, stream>>>(
        W[i], G[i], Bp[i], M[i], V[i], wh[i], bias[i], Od[i], Cd[i]);
  }
  // 2) key/value path (fo1 -> t1, fd -> vf_t, fo2(t1) -> kf_t)
  int pElems = NB * CKD * KP;
  proxy_cbr<<<dim3((pElems + 255) / 256), dim3(256), 0, stream>>>(
      proxy, wh[2], bias[2], t1, nullptr, 0, C_IN, CKD);
  proxy_cbr<<<dim3((pElems + 255) / 256), dim3(256), 0, stream>>>(
      proxy, wh[4], bias[4], nullptr, vf_t, 0, C_IN, CKD);
  proxy_cbr<<<dim3((pElems + 255) / 256), dim3(256), 0, stream>>>(
      t1, wh[3], bias[3], nullptr, kf_t, 1, CKD, CKD);
  // 3) x f32 CHW -> f16 HWC
  to_hwc_f16<<<dim3(NPIX / 64, C_IN / 64), dim3(256), 0, stream>>>(x, xh);
  // 4) fp1: xh -> H1 (bufA)
  gemm_hwc<false><<<dim3(NPIX / 64, CKD / 256), dim3(256), 0, stream>>>(
      xh, wh[0], bias[0], bufA, nullptr, C_IN, CKD);
  // 5) fp2: H1 -> Q (bufB, aliases xh which is now dead)
  gemm_hwc<false><<<dim3(NPIX / 64, CKD / 256), dim3(256), 0, stream>>>(
      bufA, wh[1], bias[1], bufB, nullptr, CKD, CKD);
  // 6) attention: Q -> ctx (bufA, H1 dead)
  attn<<<dim3(NPIX / 128), dim3(256), 0, stream>>>(bufB, kf_t, vf_t, bufA);
  // 7) fu: ctx -> out f32 CHW
  gemm_hwc<true><<<dim3(NPIX / 64, C_IN / 256), dim3(256), 0, stream>>>(
      bufA, wh[5], bias[5], nullptr, (float*)d_out, CKD, C_IN);
}
